// GNCNEncoder_9766755631465
// MI455X (gfx1250) — compile-verified, matrix operand order checked by reading, not run on hardware
//
#include <hip/hip_runtime.h>
#include <hip/hip_bf16.h>

// ---------------------------------------------------------------------------
// GNCN encoder for MI455X (gfx1250, wave32):
//   z   = L2norm_row(x @ W^T + b) * 1.8          (WMMA f16->f32 GEMM + LDS norm)
//   out = GCN-normalized 1-hop w/ self loops     (f32 atomics, bandwidth bound)
//
// W is pre-packed ONCE into f16 WMMA B-fragment register order so the hot
// GEMM loop does a single 32B fragment load for B (no per-iter f32->f16
// conversion of W, which is shared by all 3125 blocks).
// ---------------------------------------------------------------------------

#define N_NODES   50000
#define N_EDGES   1600000
#define IN_DIM    512
#define OUT_DIM   256
#define SCALE_F   1.8f
#define KTILES    (IN_DIM / 32)    // 16
#define NTILES    (OUT_DIM / 16)   // 16

typedef __attribute__((ext_vector_type(16))) _Float16 v16h;
typedef __attribute__((ext_vector_type(8)))  float    v8f;
typedef __attribute__((ext_vector_type(4)))  float    v4f;

// ---------------------------------------------------------------------------
// Kernel 0: pack W [OUT_DIM, IN_DIM] f32 into f16 B-fragments.
// Fragment (ntile, ktile) for lane l holds B[k][n] = W[n][k] with
//   n = ntile*16 + (l&15), hi = l>>4,
//   elements 0..7  <- K = ktile*32 + 8*hi + (0..7)
//   elements 8..15 <- K = ktile*32 + 16 + 8*hi + (0..7)
// Stored flat: Wh[((ntile*KTILES + ktile)*32 + lane)*16 + e].
// One wave per fragment; 256 fragments total (one-time 256 KB).
// ---------------------------------------------------------------------------
__global__ __launch_bounds__(32) void pack_w_kernel(
    const float* __restrict__ W, _Float16* __restrict__ Wh)
{
    const int tile  = blockIdx.x;          // ntile*KTILES + ktile
    const int lane  = threadIdx.x;
    const int ntile = tile >> 4;
    const int ktile = tile & 15;
    const int l15   = lane & 15;
    const int hi    = lane >> 4;

    const float* wrow = W + (size_t)(ntile * 16 + l15) * IN_DIM;
    const int ka = ktile * 32 + 8 * hi;
    const int kb = ka + 16;

    _Float16* dst = Wh + ((size_t)tile * 32 + lane) * 16;
    #pragma unroll
    for (int i = 0; i < 8; ++i) {
        dst[i]     = (_Float16)wrow[ka + i];
        dst[8 + i] = (_Float16)wrow[kb + i];
    }
}

// ---------------------------------------------------------------------------
// Kernel 1: fused GEMM + bias + row L2-normalize * SCALE.
// Block = 512 threads = 16 waves. Block b owns rows [16b, 16b+16).
// Wave w computes the 16x16 tile for columns [16w, 16w+16) via 16 chained
// v_wmma_f32_16x16x32_f16 over K=512 (unroll 2 -> WMMA pairs co-execute with
// the A-side cvt VALU). Grid exact (50000/16=3125) => EXEC all-1s for WMMA.
// ---------------------------------------------------------------------------
__global__ __launch_bounds__(512) void gemm_norm_kernel(
    const float*    __restrict__ x,      // [N_NODES, IN_DIM]
    const _Float16* __restrict__ Wh,     // packed B fragments
    const float*    __restrict__ bias,   // [OUT_DIM]
    float*          __restrict__ z)      // [N_NODES, OUT_DIM]
{
    __shared__ float zs[16][OUT_DIM];   // 16 KB tile stripe
    __shared__ float rn[16];            // per-row scale

    const int tid   = threadIdx.x;
    const int wave  = tid >> 5;
    const int lane  = tid & 31;
    const int l15   = lane & 15;
    const int hi    = lane >> 4;       // K-half selector per ISA 16-bit layout
    const int mbase = blockIdx.x * 16;
    const int nbase = wave * 16;

    const float*    xrow = x + (size_t)(mbase + l15) * IN_DIM;       // A row
    const _Float16* wfr  = Wh + (((size_t)wave * KTILES) * 32 + lane) * 16;

    v8f acc = {};
    #pragma unroll 2
    for (int kt = 0; kt < KTILES; ++kt) {
        const int kk = kt * 32;
        const int ka = kk + 8 * hi;
        const int kb = kk + 16 + 8 * hi;

        v4f a0 = *(const v4f*)(xrow + ka);
        v4f a1 = *(const v4f*)(xrow + ka + 4);
        v4f a2 = *(const v4f*)(xrow + kb);
        v4f a3 = *(const v4f*)(xrow + kb + 4);

        v16h bf = *(const v16h*)(wfr + (size_t)kt * 32 * 16);  // 2 x b128

        v16h af;
        #pragma unroll
        for (int i = 0; i < 4; ++i) {
            af[i]      = (_Float16)a0[i];
            af[4 + i]  = (_Float16)a1[i];
            af[8 + i]  = (_Float16)a2[i];
            af[12 + i] = (_Float16)a3[i];
        }

        // D = A x B + C ; (neg_a, A, neg_b, B, c_mod, C, reuse_a, reuse_b)
        acc = __builtin_amdgcn_wmma_f32_16x16x32_f16(
            false, af, false, bf, (short)0, acc, false, false);
    }

    // C/D layout: VGPR r -> row (r + 8*hi), column = nbase + l15.
    const float bv = bias[nbase + l15];
    #pragma unroll
    for (int r = 0; r < 8; ++r) {
        zs[r + 8 * hi][nbase + l15] = acc[r] + bv;
    }
    __syncthreads();

    // Row L2 norm: wave w reduces row w (256 elems, 8 per lane, stride 32).
    {
        const int row = wave;
        float s = 0.0f;
        #pragma unroll
        for (int i = 0; i < 8; ++i) {
            float v = zs[row][lane + 32 * i];
            s += v * v;
        }
        #pragma unroll
        for (int off = 16; off > 0; off >>= 1)
            s += __shfl_xor(s, off, 32);
        if (lane == 0)
            rn[row] = SCALE_F / fmaxf(sqrtf(s), 1e-12f);
    }
    __syncthreads();

    // Write normalized stripe: 4096 elems / 512 threads = 8 each.
    #pragma unroll
    for (int i = 0; i < 8; ++i) {
        int idx = tid + 512 * i;
        int r = idx >> 8;
        int c = idx & (OUT_DIM - 1);
        z[(size_t)(mbase + r) * OUT_DIM + c] = zs[r][c] * rn[r];
    }
}

// ---------------------------------------------------------------------------
// Kernel 2: degree init (self loop => deg starts at 1).
// ---------------------------------------------------------------------------
__global__ void deg_init_kernel(float* __restrict__ deg)
{
    int n = blockIdx.x * blockDim.x + threadIdx.x;
    if (n < N_NODES) deg[n] = 1.0f;
}

// ---------------------------------------------------------------------------
// Kernel 3: degree accumulation over edge destinations.
// ---------------------------------------------------------------------------
__global__ void deg_accum_kernel(const int* __restrict__ dst, float* __restrict__ deg)
{
    int e = blockIdx.x * blockDim.x + threadIdx.x;
    if (e < N_EDGES) atomicAdd(&deg[dst[e]], 1.0f);
}

// ---------------------------------------------------------------------------
// Kernel 4: dis = rsqrt(deg), in place.
// ---------------------------------------------------------------------------
__global__ void rsqrt_kernel(float* __restrict__ deg)
{
    int n = blockIdx.x * blockDim.x + threadIdx.x;
    if (n < N_NODES) deg[n] = rsqrtf(deg[n]);
}

// ---------------------------------------------------------------------------
// Kernel 5: self-loop term initializes out: out[n,:] = dis[n]^2 * z[n,:].
// Block n handles row n (256 threads = 256 cols). Also clears 0xAA poison.
// ---------------------------------------------------------------------------
__global__ __launch_bounds__(256) void self_init_kernel(
    const float* __restrict__ z, const float* __restrict__ dis,
    float* __restrict__ out)
{
    int n = blockIdx.x;
    int c = threadIdx.x;
    float d = dis[n];
    out[(size_t)n * OUT_DIM + c] = d * d * z[(size_t)n * OUT_DIM + c];
}

// ---------------------------------------------------------------------------
// Kernel 6: edge scatter. One wave per edge; lane handles 8 columns at
// stride 32 (fully coalesced). out[dst,:] += dis[src]*dis[dst]*z[src,:].
// ---------------------------------------------------------------------------
__global__ __launch_bounds__(256) void edge_scatter_kernel(
    const int* __restrict__ src, const int* __restrict__ dst,
    const float* __restrict__ z, const float* __restrict__ dis,
    float* __restrict__ out)
{
    int w = blockIdx.x * (blockDim.x >> 5) + (threadIdx.x >> 5);
    if (w >= N_EDGES) return;
    int lane = threadIdx.x & 31;

    int s = src[w];
    int d = dst[w];
    float coeff = dis[s] * dis[d];

    const float* zr = z + (size_t)s * OUT_DIM;
    float* orow = out + (size_t)d * OUT_DIM;

    #pragma unroll
    for (int i = 0; i < 8; ++i) {
        int c = lane + 32 * i;
        atomicAdd(&orow[c], coeff * zr[c]);
    }
}

// ---------------------------------------------------------------------------
// Launch
// ---------------------------------------------------------------------------
extern "C" void kernel_launch(void* const* d_in, const int* in_sizes, int n_in,
                              void* d_out, int out_size, void* d_ws, size_t ws_size,
                              hipStream_t stream)
{
    (void)in_sizes; (void)n_in; (void)out_size; (void)ws_size;

    const float* x    = (const float*)d_in[0];   // [50000, 512]
    const float* W    = (const float*)d_in[1];   // [256, 512]
    const float* bias = (const float*)d_in[2];   // [256]
    const int*   ei   = (const int*)d_in[3];     // [2, 1600000] (int32)
    const int*   src  = ei;
    const int*   dstp = ei + N_EDGES;

    float* out = (float*)d_out;                  // [50000, 256]

    // Workspace: z [50000*256] f32, dis [50000] f32, Wh [256*512] f16 (~51.7 MB).
    float*    z   = (float*)d_ws;
    float*    dis = z + (size_t)N_NODES * OUT_DIM;
    _Float16* Wh  = (_Float16*)(dis + N_NODES);

    // 0) one-time pack of W into f16 B-fragment order (256 fragments, 1 wave ea.)
    pack_w_kernel<<<NTILES * KTILES, 32, 0, stream>>>(W, Wh);

    // 1) WMMA GEMM + bias + row-normalize * 1.8
    gemm_norm_kernel<<<N_NODES / 16, 512, 0, stream>>>(x, Wh, bias, z);

    // 2-4) degree -> dis = rsqrt(deg)
    deg_init_kernel<<<(N_NODES + 255) / 256, 256, 0, stream>>>(dis);
    deg_accum_kernel<<<(N_EDGES + 255) / 256, 256, 0, stream>>>(dstp, dis);
    rsqrt_kernel<<<(N_NODES + 255) / 256, 256, 0, stream>>>(dis);

    // 5) self-loop term initializes out
    self_init_kernel<<<N_NODES, 256, 0, stream>>>(z, dis, out);

    // 6) edge scatter: 1 wave per edge, 8 waves per block
    edge_scatter_kernel<<<(N_EDGES + 7) / 8, 256, 0, stream>>>(src, dstp, z, dis, out);
}
